// TNEP_73117523247331
// MI455X (gfx1250) — matrix operand order repaired and checked by reading, not compiled, not run on hardware
//
#include <hip/hip_runtime.h>
#include <hip/hip_bf16.h>

typedef __attribute__((ext_vector_type(16))) _Float16 v16h;
typedef __attribute__((ext_vector_type(8)))  float    v8f;

#define QDIM 64
#define HDIM 64
#define MNBR 64
#define NTYPES 4

// ---------------------------------------------------------------------------
// Native CDNA5 tanh (v_tanh_f32) with branch-free fallback.
__device__ inline float fast_tanh(float x) {
#if __has_builtin(__builtin_amdgcn_tanhf)
    return __builtin_amdgcn_tanhf(x);
#else
    float e = __expf(2.0f * x);
    return 1.0f - 2.0f / (e + 1.0f);
#endif
}

// ---------------------------------------------------------------------------
// A-fragment loader: 16-bit A matrix 16x32 (MxK), per ISA 7.12.2.
// lane L: m = L&15, half = L>>4. VGPR v<4 holds K = 2v,2v+1 (+8*half);
// VGPR v>=4 holds K = 16+2(v-4), +1 (+8*half).
__device__ inline v16h load_fragA(const _Float16* __restrict__ s, int lane, int koff) {
    int m = lane & 15, half = lane >> 4;
    v16h a;
#pragma unroll
    for (int v = 0; v < 8; ++v) {
        int kb = (v < 4 ? 2 * v : 16 + 2 * (v - 4)) + 8 * half + koff;
        a[2 * v]     = s[m * QDIM + kb];
        a[2 * v + 1] = s[m * QDIM + kb + 1];
    }
    return a;
}

// B-fragment loader from LDS: 16-bit B matrix 32x16 (KxN).
// lane L: n = L&15, half = L>>4; VGPR v<4 holds K = 2v,2v+1 (+8*half); v>=4 holds K = 16+2(v-4),+1 (+8*half) (+koff).
// Element B[k][n] = src[k*ld_k + n*ld_n].
__device__ inline v16h load_fragB(const _Float16* __restrict__ src, int lane, int koff,
                                  int ld_k, int ld_n) {
    int n = lane & 15, half = lane >> 4;
    v16h b;
#pragma unroll
    for (int v = 0; v < 8; ++v) {
        int k = koff + (v < 4 ? 2 * v : 16 + 2 * (v - 4)) + 8 * half;
        b[2 * v]     = src[k * ld_k + n * ld_n];
        b[2 * v + 1] = src[(k + 1) * ld_k + n * ld_n];
    }
    return b;
}

// ---------------------------------------------------------------------------
// Stage 1: per-atom MLP chain rule -> de_dq[N][Q].
// One wave per 16-atom tile. For each of the 4 types:
//   H   = tanh(desc[16x64] @ W0_t[64x64] + b0_t)    (8 WMMAs)
//   G   = (1 - H^2) * W1_t
//   dEq = G[16x64] @ W0_t^T[64x64]                  (8 WMMAs)
// rows whose Z matches the type are written out.
__global__ __launch_bounds__(32) void mlp_dedq_kernel(
    const float* __restrict__ desc,   // [N,Q]
    const float* __restrict__ W0,     // [T,Q,H]
    const float* __restrict__ b0v,    // [T,H]
    const float* __restrict__ W1,     // [T,H]
    const int*   __restrict__ Z,      // [N]
    float* __restrict__ de_dq,        // [N,Q]
    int N) {
    __shared__ _Float16 s_desc[16 * QDIM];   // 2 KB
    __shared__ _Float16 s_g[16 * HDIM];      // 2 KB
    __shared__ _Float16 s_w0[QDIM * HDIM];   // 8 KB, current type's W0
    __shared__ int      s_z[16];

    const int lane = threadIdx.x;
    const int i0   = blockIdx.x * 16;
    const int n16  = lane & 15;
    const int half = lane >> 4;

    for (int idx = lane; idx < 16 * QDIM; idx += 32) {
        int row = i0 + (idx / QDIM);
        s_desc[idx] = (_Float16)(row < N ? desc[(size_t)i0 * QDIM + idx] : 0.0f);
    }
    if (lane < 16) s_z[lane] = (i0 + lane < N) ? Z[i0 + lane] : -1;
    __syncthreads();

    const v16h a0 = load_fragA(s_desc, lane, 0);
    const v16h a1 = load_fragA(s_desc, lane, 32);

    for (int t = 0; t < NTYPES; ++t) {
        const float* W0t = W0 + (size_t)t * QDIM * HDIM;
        __syncthreads();  // protect s_w0 / s_g reuse across t iterations
        for (int idx = lane; idx < QDIM * HDIM; idx += 32)
            s_w0[idx] = (_Float16)W0t[idx];
        __syncthreads();

        // ---- matmul 1: H columns nt*16..nt*16+15 ----
#pragma unroll
        for (int nt = 0; nt < 4; ++nt) {
            v16h bA = load_fragB(s_w0 + nt * 16, lane, 0,  HDIM, 1);
            v16h bB = load_fragB(s_w0 + nt * 16, lane, 32, HDIM, 1);
            float bias = b0v[t * HDIM + nt * 16 + n16];
            v8f c;
#pragma unroll
            for (int r = 0; r < 8; ++r) c[r] = bias;
            c = __builtin_amdgcn_wmma_f32_16x16x32_f16(false, a0, false, bA,
                                                       (short)0, c, false, false);
            c = __builtin_amdgcn_wmma_f32_16x16x32_f16(false, a1, false, bB,
                                                       (short)0, c, false, false);
            float w1v = W1[t * HDIM + nt * 16 + n16];
#pragma unroll
            for (int r = 0; r < 8; ++r) {
                float hv = fast_tanh(c[r]);
                float gv = (1.0f - hv * hv) * w1v;
                s_g[(r + 8 * half) * HDIM + nt * 16 + n16] = (_Float16)gv;
            }
        }
        __syncthreads();

        // ---- matmul 2: de_dq = G @ W0_t^T ----
        const v16h g0 = load_fragA(s_g, lane, 0);
        const v16h g1 = load_fragA(s_g, lane, 32);
#pragma unroll
        for (int nt = 0; nt < 4; ++nt) {
            // B2[k=h][n=q] = W0t[(nt*16+n)*HDIM + k]
            v16h bA = load_fragB(s_w0 + nt * 16 * HDIM, lane, 0,  1, HDIM);
            v16h bB = load_fragB(s_w0 + nt * 16 * HDIM, lane, 32, 1, HDIM);
            v8f c;
#pragma unroll
            for (int r = 0; r < 8; ++r) c[r] = 0.0f;
            c = __builtin_amdgcn_wmma_f32_16x16x32_f16(false, g0, false, bA,
                                                       (short)0, c, false, false);
            c = __builtin_amdgcn_wmma_f32_16x16x32_f16(false, g1, false, bB,
                                                       (short)0, c, false, false);
#pragma unroll
            for (int r = 0; r < 8; ++r) {
                int row = r + 8 * half;
                if (s_z[row] == t)
                    de_dq[(size_t)(i0 + row) * QDIM + nt * 16 + n16] = c[r];
            }
        }
    }
}

// ---------------------------------------------------------------------------
// Stage 2: fused r2 * force contraction. One block per atom i (192 threads).
// dipole_c -= sum_q de_dq[i,q] * sum_m r2[i,m] * grad[i,m,c,q]
// Thread (mg, c, q4) streams float4 columns -> global_load_b128, fully
// coalesced 512B/wave over the 201 MB gradients array (~8.7us roofline).
__global__ __launch_bounds__(192) void dipole_partial_kernel(
    const float* __restrict__ grads,  // [N,M,3,Q]
    const float* __restrict__ pos,    // [N,3]
    const float* __restrict__ box,    // [3,3]
    const int*   __restrict__ gidx,   // [N,M]
    const float* __restrict__ de_dq,  // [N,Q]
    float* __restrict__ partial,      // [N,3]
    int N) {
    __shared__ float s_r2[MNBR];
    __shared__ float s_dq[QDIM];
    __shared__ float s_red[192];

    const int i   = blockIdx.x;
    const int tid = threadIdx.x;

    if (tid < QDIM) s_dq[tid] = de_dq[(size_t)i * QDIM + tid];

    if (tid < MNBR) {
        // general 3x3 inverse (adjugate); box is tiny, redundant per thread ok
        float a = box[0], b = box[1], c = box[2];
        float d = box[3], e = box[4], f = box[5];
        float g = box[6], h = box[7], k = box[8];
        float A =  e * k - f * h, B = c * h - b * k, C = b * f - c * e;
        float D =  f * g - d * k, E = a * k - c * g, F = c * d - a * f;
        float G =  d * h - e * g, H = b * g - a * h, K = a * e - b * d;
        float inv_det = 1.0f / (a * A + b * D + c * G);

        int j = gidx[(size_t)i * MNBR + tid];
        float dx = pos[j * 3 + 0] - pos[i * 3 + 0];
        float dy = pos[j * 3 + 1] - pos[i * 3 + 1];
        float dz = pos[j * 3 + 2] - pos[i * 3 + 2];
        float sx = (dx * A + dy * D + dz * G) * inv_det;
        float sy = (dx * B + dy * E + dz * H) * inv_det;
        float sz = (dx * C + dy * F + dz * K) * inv_det;
        sx -= rintf(sx); sy -= rintf(sy); sz -= rintf(sz);
        float wx = sx * a + sy * d + sz * g;
        float wy = sx * b + sy * e + sz * h;
        float wz = sx * c + sy * f + sz * k;
        s_r2[tid] = wx * wx + wy * wy + wz * wz;
    }
    __syncthreads();

    // thread -> (m-phase mg 0..3, c 0..2, q-quad 0..15)
    const int mg = tid / 48;
    const int r  = tid - mg * 48;
    const int c  = r >> 4;
    const int q4 = (r & 15) * 4;
    const float4* gb = (const float4*)(grads + (size_t)i * MNBR * 3 * QDIM +
                                       c * QDIM + q4);
    const int mstride4 = (3 * QDIM) / 4;  // 48 float4 per m
    float4 acc = make_float4(0.f, 0.f, 0.f, 0.f);
#pragma unroll 4
    for (int m = mg; m < MNBR; m += 4) {
        float4 gv = gb[(size_t)m * mstride4];
        float r2 = s_r2[m];
        acc.x += r2 * gv.x;
        acc.y += r2 * gv.y;
        acc.z += r2 * gv.z;
        acc.w += r2 * gv.w;
    }
    float v = acc.x * s_dq[q4] + acc.y * s_dq[q4 + 1] +
              acc.z * s_dq[q4 + 2] + acc.w * s_dq[q4 + 3];
    s_red[c * 64 + mg * 16 + (r & 15)] = v;
    __syncthreads();

    // fixed-order tree reduction over 64 entries per c (deterministic)
    const int rc = tid >> 6, rq = tid & 63;
#pragma unroll
    for (int off = 32; off >= 1; off >>= 1) {
        if (rq < off) s_red[rc * 64 + rq] += s_red[rc * 64 + rq + off];
        __syncthreads();
    }
    if (rq == 0) partial[(size_t)i * 3 + rc] = -s_red[rc * 64];
}

// ---------------------------------------------------------------------------
// Stage 3: deterministic fixed-order reduction of per-atom partials -> [3].
__global__ __launch_bounds__(256) void final_reduce_kernel(
    const float* __restrict__ partial, float* __restrict__ out, int N) {
    __shared__ float s[256];
    for (int c = 0; c < 3; ++c) {
        float acc = 0.0f;
        for (int i = threadIdx.x; i < N; i += 256)
            acc += partial[(size_t)i * 3 + c];
        s[threadIdx.x] = acc;
        __syncthreads();
#pragma unroll
        for (int off = 128; off >= 1; off >>= 1) {
            if ((int)threadIdx.x < off) s[threadIdx.x] += s[threadIdx.x + off];
            __syncthreads();
        }
        if (threadIdx.x == 0) out[c] = s[0];
        __syncthreads();
    }
}

// ---------------------------------------------------------------------------
extern "C" void kernel_launch(void* const* d_in, const int* in_sizes, int n_in,
                              void* d_out, int out_size, void* d_ws, size_t ws_size,
                              hipStream_t stream) {
    const float* desc  = (const float*)d_in[0];   // [N,Q]
    const float* grads = (const float*)d_in[1];   // [N,M,3,Q]
    const float* pos   = (const float*)d_in[2];   // [N,3]
    const float* box   = (const float*)d_in[3];   // [3,3]
    const float* W0    = (const float*)d_in[4];   // [T,Q,H]
    const float* b0    = (const float*)d_in[5];   // [T,H]
    const float* W1    = (const float*)d_in[6];   // [T,H]
    const int*   Z     = (const int*)d_in[7];     // [N]
    const int*   gidx  = (const int*)d_in[8];     // [N,M]

    const int N = in_sizes[0] / QDIM;

    float* de_dq   = (float*)d_ws;                 // N*Q floats (1 MB)
    float* partial = de_dq + (size_t)N * QDIM;     // N*3 floats

    mlp_dedq_kernel<<<(N + 15) / 16, 32, 0, stream>>>(desc, W0, b0, W1, Z, de_dq, N);
    dipole_partial_kernel<<<N, 192, 0, stream>>>(grads, pos, box, gidx, de_dq,
                                                 partial, N);
    final_reduce_kernel<<<1, 256, 0, stream>>>(partial, (float*)d_out, N);
}
